// GCN2_48524540510792
// MI455X (gfx1250) — compile-verified, matrix operand order checked by reading, not run on hardware
//
#include <hip/hip_runtime.h>
#include <hip/hip_bf16.h>
#include <stdint.h>

// ---------------------------------------------------------------------------
// Problem constants (match the reference)
// ---------------------------------------------------------------------------
#define NN     20000
#define E1N    160000
#define E2N    480000
#define F_IN   512
#define H1     1024
#define H2     512
#define C_OUT  128

typedef __attribute__((ext_vector_type(16))) __bf16 v16bf;
typedef __attribute__((ext_vector_type(8)))  float  v8f;

static __device__ __forceinline__ unsigned short f32_to_bf16_rne(float f) {
    unsigned int x = __float_as_uint(f);
    unsigned int r = x + 0x7FFFu + ((x >> 16) & 1u);   // round-to-nearest-even
    return (unsigned short)(r >> 16);
}

// ---------------------------------------------------------------------------
// Cast f32 -> bf16 with optional column concatenation into a fused weight
// matrix:  dst[k*ldd + coff + n] = bf16(src[k*nsrc + n])
// ---------------------------------------------------------------------------
__global__ void cast_concat_bf16(const float* __restrict__ src,
                                 unsigned short* __restrict__ dst,
                                 int k_rows, int nsrc, int ldd, int coff) {
    int idx = blockIdx.x * blockDim.x + threadIdx.x;
    int total = k_rows * nsrc;
    if (idx >= total) return;
    int k = idx / nsrc;
    int n = idx - k * nsrc;
    dst[(size_t)k * ldd + coff + n] = f32_to_bf16_rne(src[idx]);
}

// ---------------------------------------------------------------------------
// bf16 WMMA GEMM:  C[M,N] = A[M,K] * B[K,N]     (A,B bf16 row-major, C f32)
// 256 threads = 8 waves; block tile 128x128; K step 32.
// Register double-buffered staging: next K-tile's global_load_b128s are
// issued before the WMMA phase of the current tile.
// LDS rows padded to 20 dwords (80B) -> 16B-aligned for ds_store_b128.
// ---------------------------------------------------------------------------
#define TM 128
#define TN 128
#define TK 32
#define LDSW 20

__global__ __launch_bounds__(256)
void gemm_bf16_wmma(const unsigned short* __restrict__ A,
                    const unsigned short* __restrict__ B,
                    float* __restrict__ C,
                    int M, int N, int K) {
    __shared__ uint32_t As[TM][LDSW];   // [row][kdword], 16 used
    __shared__ uint32_t Bs[TN][LDSW];   // transposed: [col][kdword]

    const int tid   = threadIdx.x;
    const int lane  = tid & 31;
    const int wave  = tid >> 5;      // 0..7
    const int wm    = wave >> 1;     // 0..3  (row group of 32)
    const int wn    = wave & 1;      // 0..1  (col group of 64)
    const int l16   = lane & 15;
    const int khalf = lane >> 4;     // selects K interleave half

    const int block_row = blockIdx.x * TM;
    const int block_col = blockIdx.y * TN;

    // --- staging thread mapping (fixed per thread) ---
    // A: 2 iterations, each a b128 load: row ar[i], dword col ac4[i]
    const int ar0 = tid >> 2;                    // 0..63
    const int ar1 = ar0 + 64;                    // 64..127
    const int ac4 = (tid & 3) * 4;               // dword col {0,4,8,12}
    const int agr0 = min(block_row + ar0, M - 1);// clamped: branchless staging
    const int agr1 = min(block_row + ar1, M - 1);
    // B: one K-pair x 8 columns per thread
    const int gkp = tid >> 4;                    // 0..15 (pair of K rows)
    const int bn0 = (tid & 15) * 8;              // 0..120

    v8f acc[2][4];
#pragma unroll
    for (int mt = 0; mt < 2; ++mt)
#pragma unroll
        for (int nt = 0; nt < 4; ++nt)
            acc[mt][nt] = (v8f){0.f, 0.f, 0.f, 0.f, 0.f, 0.f, 0.f, 0.f};

    union Frag { v16bf v; uint32_t u[8]; };

    uint4 aReg0, aReg1, bReg0, bReg1;

    // ---- prologue: load first K-tile into registers ----
    {
        const unsigned short* a0 = A + (size_t)agr0 * K + ac4 * 2;
        const unsigned short* a1 = A + (size_t)agr1 * K + ac4 * 2;
        aReg0 = *(const uint4*)a0;
        aReg1 = *(const uint4*)a1;
        const unsigned short* bp = B + (size_t)(gkp * 2) * N + block_col + bn0;
        bReg0 = *(const uint4*)bp;
        bReg1 = *(const uint4*)(bp + N);
    }

    for (int kb = 0; kb < K; kb += TK) {
        // ---- commit staged registers to LDS ----
        *(uint4*)&As[ar0][ac4] = aReg0;
        *(uint4*)&As[ar1][ac4] = aReg1;
        {
            const uint32_t* w0 = (const uint32_t*)&bReg0;
            const uint32_t* w1 = (const uint32_t*)&bReg1;
#pragma unroll
            for (int j = 0; j < 8; ++j) {
                uint32_t e0 = (j & 1) ? (w0[j >> 1] >> 16) : (w0[j >> 1] & 0xFFFFu);
                uint32_t e1 = (j & 1) ? (w1[j >> 1] >> 16) : (w1[j >> 1] & 0xFFFFu);
                Bs[bn0 + j][gkp] = e0 | (e1 << 16);   // pack (K, K+1) per column
            }
        }
        __syncthreads();

        // ---- issue next K-tile's global loads (overlap with WMMAs) ----
        int kn = kb + TK;
        if (kn < K) {
            const unsigned short* a0 = A + (size_t)agr0 * K + kn + ac4 * 2;
            const unsigned short* a1 = A + (size_t)agr1 * K + kn + ac4 * 2;
            aReg0 = *(const uint4*)a0;
            aReg1 = *(const uint4*)a1;
            const unsigned short* bp = B + (size_t)(kn + gkp * 2) * N + block_col + bn0;
            bReg0 = *(const uint4*)bp;
            bReg1 = *(const uint4*)(bp + N);
        }

        // ---- gather fragments (ISA 16-bit A/B layout) ----
        Frag a[2], b[4];
#pragma unroll
        for (int mt = 0; mt < 2; ++mt) {
            int m = wm * 32 + mt * 16 + l16;
#pragma unroll
            for (int d = 0; d < 4; ++d) {
                a[mt].u[d]     = As[m][khalf * 4 + d];       // K = half*8 + 0..7
                a[mt].u[4 + d] = As[m][8 + khalf * 4 + d];   // K = 16 + half*8 + 0..7
            }
        }
#pragma unroll
        for (int nt = 0; nt < 4; ++nt) {
            int n = wn * 64 + nt * 16 + l16;
#pragma unroll
            for (int d = 0; d < 4; ++d) {
                b[nt].u[d]     = Bs[n][khalf * 4 + d];
                b[nt].u[4 + d] = Bs[n][8 + khalf * 4 + d];
            }
        }

        // ---- 8 WMMAs per K-step ----
#pragma unroll
        for (int mt = 0; mt < 2; ++mt)
#pragma unroll
            for (int nt = 0; nt < 4; ++nt)
                acc[mt][nt] = __builtin_amdgcn_wmma_f32_16x16x32_bf16(
                    false, a[mt].v, false, b[nt].v,
                    (short)0, acc[mt][nt], false, false);
        __syncthreads();
    }

    // ---- store: C/D layout => VGPR r holds row (r + khalf*8), col = l16 ----
#pragma unroll
    for (int mt = 0; mt < 2; ++mt)
#pragma unroll
        for (int nt = 0; nt < 4; ++nt) {
            int gc = block_col + wn * 64 + nt * 16 + l16;
#pragma unroll
            for (int r = 0; r < 8; ++r) {
                int gr = block_row + wm * 32 + mt * 16 + khalf * 8 + r;
                if (gr < M)
                    C[(size_t)gr * N + gc] = acc[mt][nt][r];
            }
        }
}

// ---------------------------------------------------------------------------
// Edge aggregation: one wave32 per edge.
//   out[dst[e], :] += ew[e] * S[src[e], coff : coff+C]
// float4 gathers (mostly L2 hits) + native f32 atomic scatters.
// ---------------------------------------------------------------------------
__global__ __launch_bounds__(256)
void edge_aggregate(const int* __restrict__ src, const int* __restrict__ dst,
                    const float* __restrict__ ew, int E,
                    const float* __restrict__ S, int ldS, int coff,
                    float* __restrict__ out, int C) {
    int wave = blockIdx.x * (blockDim.x >> 5) + (threadIdx.x >> 5);
    int lane = threadIdx.x & 31;
    if (wave >= E) return;
    int s = src[wave];
    int d = dst[wave];
    float w = ew[wave];
    const float* __restrict__ srow = S + (size_t)s * ldS + coff;
    float* __restrict__ drow = out + (size_t)d * C;
    for (int c = lane * 4; c < C; c += 128) {
        float4 v = *(const float4*)(srow + c);
        __hip_atomic_fetch_add(&drow[c + 0], v.x * w, __ATOMIC_RELAXED, __HIP_MEMORY_SCOPE_AGENT);
        __hip_atomic_fetch_add(&drow[c + 1], v.y * w, __ATOMIC_RELAXED, __HIP_MEMORY_SCOPE_AGENT);
        __hip_atomic_fetch_add(&drow[c + 2], v.z * w, __ATOMIC_RELAXED, __HIP_MEMORY_SCOPE_AGENT);
        __hip_atomic_fetch_add(&drow[c + 3], v.w * w, __ATOMIC_RELAXED, __HIP_MEMORY_SCOPE_AGENT);
    }
}

// ---------------------------------------------------------------------------
// Layer epilogue: per-channel softmax mix of the two ELU'd GCN branches plus
// the skip linear columns of S; emits bf16 activation for the next GEMM
// and/or the final f32 output.
// ---------------------------------------------------------------------------
__global__ __launch_bounds__(256)
void layer_epilogue(const float* __restrict__ agg1, const float* __restrict__ agg2,
                    const float* __restrict__ S, int ldS, int coff,
                    const float* __restrict__ bg1, const float* __restrict__ bg2,
                    const float* __restrict__ blin,
                    const float* __restrict__ amix,    // [C,2] mixing logits
                    unsigned short* __restrict__ act_bf16,   // may be null
                    float* __restrict__ out_f32,             // may be null
                    int rows, int C) {
    int idx = blockIdx.x * blockDim.x + threadIdx.x;
    int total = rows * C;
    if (idx >= total) return;
    int n = idx / C;
    int c = idx - n * C;

    float l0 = amix[c * 2 + 0], l1 = amix[c * 2 + 1];
    float mx = fmaxf(l0, l1);
    float e0 = __expf(l0 - mx), e1 = __expf(l1 - mx);
    float inv = 1.0f / (e0 + e1);
    float w0 = e0 * inv, w1 = e1 * inv;

    float x1 = agg1[idx] + bg1[c];
    float x2 = agg2[idx] + bg2[c];
    x1 = (x1 > 0.0f) ? x1 : (__expf(x1) - 1.0f);   // ELU
    x2 = (x2 > 0.0f) ? x2 : (__expf(x2) - 1.0f);

    float lin = S[(size_t)n * ldS + coff + c] + blin[c];
    float r = w0 * x1 + w1 * x2 + lin;

    if (act_bf16) act_bf16[idx] = f32_to_bf16_rne(r);
    if (out_f32)  out_f32[idx]  = r;
}

// ---------------------------------------------------------------------------
// Host-side orchestration
// ---------------------------------------------------------------------------
static inline size_t align256(size_t x) { return (x + 255) & ~(size_t)255; }

extern "C" void kernel_launch(void* const* d_in, const int* in_sizes, int n_in,
                              void* d_out, int out_size, void* d_ws, size_t ws_size,
                              hipStream_t stream) {
    (void)in_sizes; (void)n_in; (void)out_size; (void)ws_size;

    // ---- inputs (setup_inputs dict order) ----
    const float* node_feature = (const float*)d_in[0];
    const int*   src1 = (const int*)d_in[1];
    const int*   dst1 = (const int*)d_in[2];
    const float* ew1  = (const float*)d_in[3];
    const int*   src2 = (const int*)d_in[4];
    const int*   dst2 = (const int*)d_in[5];
    const float* ew2  = (const float*)d_in[6];
    const float* w_g11 = (const float*)d_in[7];   const float* b_g11 = (const float*)d_in[8];
    const float* w_g12 = (const float*)d_in[9];   const float* b_g12 = (const float*)d_in[10];
    const float* w_g21 = (const float*)d_in[11];  const float* b_g21 = (const float*)d_in[12];
    const float* w_g22 = (const float*)d_in[13];  const float* b_g22 = (const float*)d_in[14];
    const float* w_g31 = (const float*)d_in[15];  const float* b_g31 = (const float*)d_in[16];
    const float* w_g32 = (const float*)d_in[17];  const float* b_g32 = (const float*)d_in[18];
    const float* w_lin1 = (const float*)d_in[19]; const float* b_lin1 = (const float*)d_in[20];
    const float* w_lin2 = (const float*)d_in[21]; const float* b_lin2 = (const float*)d_in[22];
    const float* w_lin3 = (const float*)d_in[23]; const float* b_lin3 = (const float*)d_in[24];
    const float* a1 = (const float*)d_in[25];
    const float* a2 = (const float*)d_in[26];
    const float* a3 = (const float*)d_in[27];

    float* out = (float*)d_out;                     // [N, C_OUT] f32

    // ---- workspace carve-up (reused across layers) ----
    char* ws = (char*)d_ws;
    size_t off = 0;
    unsigned short* act = (unsigned short*)(ws + off);          // bf16 activations
    off = align256(off + (size_t)NN * H1 * sizeof(unsigned short));
    unsigned short* wc1 = (unsigned short*)(ws + off);          // [512,3072] bf16
    off = align256(off + (size_t)F_IN * (3 * H1) * sizeof(unsigned short));
    unsigned short* wc2 = (unsigned short*)(ws + off);          // [1024,1536] bf16
    off = align256(off + (size_t)H1 * (3 * H2) * sizeof(unsigned short));
    unsigned short* wc3 = (unsigned short*)(ws + off);          // [512,384] bf16
    off = align256(off + (size_t)H2 * (3 * C_OUT) * sizeof(unsigned short));
    float* S = (float*)(ws + off);                              // dense GEMM out
    off = align256(off + (size_t)NN * (3 * H1) * sizeof(float));
    float* agg1 = (float*)(ws + off);
    off = align256(off + (size_t)NN * H1 * sizeof(float));
    float* agg2 = (float*)(ws + off);
    off = align256(off + (size_t)NN * H1 * sizeof(float));

    const int THREADS = 256;
    auto blocks1d = [&](long long total) { return (unsigned)((total + THREADS - 1) / THREADS); };

    // =========================== weight fusion =============================
    cast_concat_bf16<<<blocks1d((long long)F_IN * H1), THREADS, 0, stream>>>(w_g11, wc1, F_IN, H1, 3 * H1, 0);
    cast_concat_bf16<<<blocks1d((long long)F_IN * H1), THREADS, 0, stream>>>(w_g12, wc1, F_IN, H1, 3 * H1, H1);
    cast_concat_bf16<<<blocks1d((long long)F_IN * H1), THREADS, 0, stream>>>(w_lin1, wc1, F_IN, H1, 3 * H1, 2 * H1);
    cast_concat_bf16<<<blocks1d((long long)H1 * H2), THREADS, 0, stream>>>(w_g21, wc2, H1, H2, 3 * H2, 0);
    cast_concat_bf16<<<blocks1d((long long)H1 * H2), THREADS, 0, stream>>>(w_g22, wc2, H1, H2, 3 * H2, H2);
    cast_concat_bf16<<<blocks1d((long long)H1 * H2), THREADS, 0, stream>>>(w_lin2, wc2, H1, H2, 3 * H2, 2 * H2);
    cast_concat_bf16<<<blocks1d((long long)H2 * C_OUT), THREADS, 0, stream>>>(w_g31, wc3, H2, C_OUT, 3 * C_OUT, 0);
    cast_concat_bf16<<<blocks1d((long long)H2 * C_OUT), THREADS, 0, stream>>>(w_g32, wc3, H2, C_OUT, 3 * C_OUT, C_OUT);
    cast_concat_bf16<<<blocks1d((long long)H2 * C_OUT), THREADS, 0, stream>>>(w_lin3, wc3, H2, C_OUT, 3 * C_OUT, 2 * C_OUT);

    // ============================== layer 1 ================================
    cast_concat_bf16<<<blocks1d((long long)NN * F_IN), THREADS, 0, stream>>>(node_feature, act, NN, F_IN, F_IN, 0);
    {
        dim3 grid((NN + TM - 1) / TM, (3 * H1) / TN);
        gemm_bf16_wmma<<<grid, THREADS, 0, stream>>>(act, wc1, S, NN, 3 * H1, F_IN);
    }
    hipMemsetAsync(agg1, 0, (size_t)NN * H1 * sizeof(float), stream);
    hipMemsetAsync(agg2, 0, (size_t)NN * H1 * sizeof(float), stream);
    edge_aggregate<<<(E1N + 7) / 8, THREADS, 0, stream>>>(src1, dst1, ew1, E1N, S, 3 * H1, 0,   agg1, H1);
    edge_aggregate<<<(E2N + 7) / 8, THREADS, 0, stream>>>(src2, dst2, ew2, E2N, S, 3 * H1, H1,  agg2, H1);
    layer_epilogue<<<blocks1d((long long)NN * H1), THREADS, 0, stream>>>(
        agg1, agg2, S, 3 * H1, 2 * H1, b_g11, b_g12, b_lin1, a1,
        act, (float*)nullptr, NN, H1);

    // ============================== layer 2 ================================
    {
        dim3 grid((NN + TM - 1) / TM, (3 * H2) / TN);
        gemm_bf16_wmma<<<grid, THREADS, 0, stream>>>(act, wc2, S, NN, 3 * H2, H1);
    }
    hipMemsetAsync(agg1, 0, (size_t)NN * H2 * sizeof(float), stream);
    hipMemsetAsync(agg2, 0, (size_t)NN * H2 * sizeof(float), stream);
    edge_aggregate<<<(E1N + 7) / 8, THREADS, 0, stream>>>(src1, dst1, ew1, E1N, S, 3 * H2, 0,   agg1, H2);
    edge_aggregate<<<(E2N + 7) / 8, THREADS, 0, stream>>>(src2, dst2, ew2, E2N, S, 3 * H2, H2,  agg2, H2);
    layer_epilogue<<<blocks1d((long long)NN * H2), THREADS, 0, stream>>>(
        agg1, agg2, S, 3 * H2, 2 * H2, b_g21, b_g22, b_lin2, a2,
        act, (float*)nullptr, NN, H2);

    // ============================== layer 3 ================================
    {
        dim3 grid((NN + TM - 1) / TM, (3 * C_OUT) / TN);
        gemm_bf16_wmma<<<grid, THREADS, 0, stream>>>(act, wc3, S, NN, 3 * C_OUT, H2);
    }
    hipMemsetAsync(agg1, 0, (size_t)NN * C_OUT * sizeof(float), stream);
    hipMemsetAsync(agg2, 0, (size_t)NN * C_OUT * sizeof(float), stream);
    edge_aggregate<<<(E1N + 7) / 8, THREADS, 0, stream>>>(src1, dst1, ew1, E1N, S, 3 * C_OUT, 0,     agg1, C_OUT);
    edge_aggregate<<<(E2N + 7) / 8, THREADS, 0, stream>>>(src2, dst2, ew2, E2N, S, 3 * C_OUT, C_OUT, agg2, C_OUT);
    layer_epilogue<<<blocks1d((long long)NN * C_OUT), THREADS, 0, stream>>>(
        agg1, agg2, S, 3 * C_OUT, 2 * C_OUT, b_g31, b_g32, b_lin3, a3,
        (unsigned short*)nullptr, out, NN, C_OUT);
}
